// PartNet_19344532702245
// MI455X (gfx1250) — matrix operand order repaired and checked by reading, not compiled, not run on hardware
//
#include <hip/hip_runtime.h>
#include <hip/hip_bf16.h>
#include <stdint.h>

typedef __attribute__((ext_vector_type(16))) _Float16 v16h;
typedef __attribute__((ext_vector_type(8)))  float    v8f;

#define NPARTS  8
#define HD      128
#define NL      2
#define NPOINTS 200000
#define MT      64      // rows (points) per block
#define HSTR    136     // f16 elems per LDS row (128 + 8 pad -> 272B, 16B aligned)
#define HSTRU   68      // dwords per LDS row

union FragH { v16h h; uint32_t u[8]; };
union AccF  { v8f  v; float f[8]; };
union PackH { _Float16 h[2]; uint32_t u; };

__device__ __forceinline__ float softplus_f(float v) {
    return v > 20.0f ? v : __logf(1.0f + __expf(v));
}

// ---------------------------------------------------------------- c-network
__global__ void k_cnet(const float* __restrict__ c,
                       const float* __restrict__ WB1, const float* __restrict__ bB1,
                       const float* __restrict__ WB2, const float* __restrict__ bB2,
                       const float* __restrict__ WB3, const float* __restrict__ bB3,
                       float* __restrict__ y2) {
    __shared__ float h1[HD], h2[HD];
    int j = threadIdx.x;
    float acc = bB1[j];
    #pragma unroll
    for (int i = 0; i < 16; ++i) acc += c[i] * WB1[i * HD + j];
    h1[j] = softplus_f(acc);
    __syncthreads();
    acc = bB2[j];
    for (int k = 0; k < HD; ++k) acc += h1[k] * WB2[k * HD + j];
    h2[j] = softplus_f(acc);
    __syncthreads();
    acc = bB3[j];
    for (int k = 0; k < HD; ++k) acc += h2[k] * WB3[k * HD + j];
    y2[j] = acc;
}

// --------------------------------------- fold y2 into output weights/biases
__global__ void k_vout(const float* __restrict__ WA_out, const float* __restrict__ bA_out,
                       const float* __restrict__ y2,
                       float* __restrict__ vA, float* __restrict__ bOut) {
    int n = blockIdx.x;
    int t = threadIdx.x;           // 0..383 -> (k, y)
    __shared__ float sy2[HD];
    if (t < HD) sy2[t] = y2[t];
    __syncthreads();
    int k = t / 3, yy = t - k * 3;
    const float* W = WA_out + (size_t)(n * HD + k) * (3 * HD) + yy * HD;
    float acc = 0.0f;
    for (int h = 0; h < HD; ++h) acc += W[h] * sy2[h];
    vA[n * HD * 3 + k * 3 + yy] = acc;
    if (t < 3) {
        const float* b = bA_out + n * 3 * HD + t * HD;
        float a2 = 0.0f;
        for (int h = 0; h < HD; ++h) a2 += b[h] * sy2[h];
        bOut[n * 3 + t] = a2;
    }
}

// -------- repack WA_hid (f32, [n][l][k_in][k_out]) into f16 B-fragment order
// Layout: frag = ((n*NL+l)*8 + nt)*4 + kt ; dword index = (frag*32 + lane)*8 + v
// lane: lo=lane&15 -> column nt*16+lo ; hi=lane>>4 ; pair K = kt*32 + hi*16 + 2v
__global__ void k_repack(const float* __restrict__ WA_hid, uint32_t* __restrict__ WtU) {
    int tid  = blockIdx.x * blockDim.x + threadIdx.x;   // 0..65535
    int v    = tid & 7;
    int lane = (tid >> 3) & 31;
    int frag = tid >> 8;
    int kt   = frag & 3;
    int nt   = (frag >> 2) & 7;
    int nl   = frag >> 5;                               // n*NL + l
    int lo = lane & 15, hi = lane >> 4;
    int col = nt * 16 + lo;
    int k   = kt * 32 + hi * 16 + 2 * v;
    const float* W = WA_hid + (size_t)nl * HD * HD;
    PackH p;
    p.h[0] = (_Float16)W[k * HD + col];
    p.h[1] = (_Float16)W[(k + 1) * HD + col];
    WtU[tid] = p.u;
}

// ------------------------------------------------------------- main kernel
__global__ __launch_bounds__(256) void k_main(
        const float* __restrict__ x,
        const float* __restrict__ WA_in,    // [8][3][128]
        const float* __restrict__ bA_in,    // [8][128]
        const float* __restrict__ bA_hid,   // [8][2][128]
        const uint32_t* __restrict__ WtU,   // repacked f16 fragments
        const float* __restrict__ vA,       // [8][128][3]
        const float* __restrict__ bOut,     // [8][3]
        float* __restrict__ out) {          // [N][3]
    __shared__ __align__(16) _Float16 sH[2][MT * HSTR];
    __shared__ float sX[MT * 3];
    __shared__ float sCoef[MT][NPARTS];

    const int tid  = threadIdx.x;
    const int row0 = blockIdx.x * MT;

    if (tid < MT * 3) sX[tid] = x[row0 * 3 + tid];
    if (tid < MT) {
        float z = x[(row0 + tid) * 3 + 2];
        float w[NPARTS]; float s = 0.0f;
        #pragma unroll
        for (int n = 0; n < NPARTS; ++n) {
            float a = -1.0f + 2.0f * (float)n / 7.0f;
            float d = z - a;
            w[n] = __expf(-0.5f * d * d / 0.005625f);   // b = (0.3*2/8)^2
            s += w[n];
        }
        float inv = 1.0f / s;
        sCoef[tid][0] = 1.0f;                            // part 0 coeff is 1.0
        #pragma unroll
        for (int n = 1; n < NPARTS; ++n) sCoef[tid][n] = w[n] * inv;
    }

    const int lane  = tid & 31;
    const int wv    = tid >> 5;     // 8 waves
    const int mrow  = wv & 3;       // 16-row subtile
    const int nhalf = wv >> 2;      // column half (64 cols each)
    const int lo = lane & 15;
    const int hi = lane >> 4;

    const int orow = tid / 3, oy = tid - orow * 3;   // output ownership (tid<192)
    float accY = 0.0f;

    for (int n = 0; n < NPARTS; ++n) {
        __syncthreads();   // protect sH[0] from previous part's output reads

        // ---- layer 0: K=3, VALU, write f16 activations into sH[0]
        const float* Wi = WA_in + n * 3 * HD;
        const float* bi = bA_in + n * HD;
        for (int o = tid; o < MT * HD; o += 256) {
            int r = o >> 7, col = o & 127;
            float v = bi[col]
                    + sX[r * 3 + 0] * Wi[0 * HD + col]
                    + sX[r * 3 + 1] * Wi[1 * HD + col]
                    + sX[r * 3 + 2] * Wi[2 * HD + col];
            sH[0][r * HSTR + col] = (_Float16)softplus_f(v);
        }
        __syncthreads();

        // ---- 2 hidden layers: 64x128 @ 128x128 via WMMA f16->f32
        #pragma unroll
        for (int l = 0; l < NL; ++l) {
            const uint32_t* hinU = (const uint32_t*)sH[l & 1];
            _Float16*       hout = sH[(l & 1) ^ 1];
            const float*    bh   = bA_hid + (n * NL + l) * HD;
            const uint32_t* wbase = WtU + (size_t)(n * NL + l) * 8192;

            AccF c0, c1, c2, c3;
            {
                float b0 = bh[nhalf * 64 +  0 + lo];
                float b1 = bh[nhalf * 64 + 16 + lo];
                float b2 = bh[nhalf * 64 + 32 + lo];
                float b3 = bh[nhalf * 64 + 48 + lo];
                #pragma unroll
                for (int r = 0; r < 8; ++r) {
                    c0.f[r] = b0; c1.f[r] = b1; c2.f[r] = b2; c3.f[r] = b3;
                }
            }

            #pragma unroll
            for (int kt = 0; kt < 4; ++kt) {
                // A fragment from LDS (rows mrow*16+lo, K contiguous: k = kt*32 + hi*16 + s)
                FragH a;
                const uint32_t* pA = hinU + (mrow * 16 + lo) * HSTRU + kt * 16 + hi * 8;
                uint4 qa0 = *(const uint4*)pA;
                uint4 qa1 = *(const uint4*)(pA + 4);
                a.u[0] = qa0.x; a.u[1] = qa0.y; a.u[2] = qa0.z; a.u[3] = qa0.w;
                a.u[4] = qa1.x; a.u[5] = qa1.y; a.u[6] = qa1.z; a.u[7] = qa1.w;

                #pragma unroll
                for (int t = 0; t < 4; ++t) {
                    FragH b;
                    const uint32_t* pB =
                        wbase + ((size_t)(((nhalf * 4 + t) * 4 + kt) * 32 + lane)) * 8;
                    uint4 qb0 = *(const uint4*)pB;
                    uint4 qb1 = *(const uint4*)(pB + 4);
                    b.u[0] = qb0.x; b.u[1] = qb0.y; b.u[2] = qb0.z; b.u[3] = qb0.w;
                    b.u[4] = qb1.x; b.u[5] = qb1.y; b.u[6] = qb1.z; b.u[7] = qb1.w;

                    AccF* cc = (t == 0) ? &c0 : (t == 1) ? &c1 : (t == 2) ? &c2 : &c3;
                    cc->v = __builtin_amdgcn_wmma_f32_16x16x32_f16(
                        false, a.h, false, b.h, (short)0, cc->v, false, false);
                }
            }

            // softplus + f16 store to the other buffer
            #pragma unroll
            for (int t = 0; t < 4; ++t) {
                AccF* cc = (t == 0) ? &c0 : (t == 1) ? &c1 : (t == 2) ? &c2 : &c3;
                int col = nhalf * 64 + t * 16 + lo;
                #pragma unroll
                for (int r = 0; r < 8; ++r) {
                    int m = mrow * 16 + hi * 8 + r;
                    hout[m * HSTR + col] = (_Float16)softplus_f(cc->f[r]);
                }
            }
            __syncthreads();
        }

        // ---- folded output layer: y_part = h @ vA[n] + bOut[n]; blend by coef
        if (tid < MT * 3) {
            const float*    vp   = vA + n * HD * 3 + oy;
            const _Float16* hrow = sH[0] + orow * HSTR;
            float acc = bOut[n * 3 + oy];
            #pragma unroll 8
            for (int k = 0; k < HD; ++k) acc += (float)hrow[k] * vp[k * 3];
            accY += sCoef[orow][n] * acc;
        }
    }

    if (tid < MT * 3) {
        int row = row0 + orow;
        if (row < NPOINTS) out[row * 3 + oy] = accY;
    }
}

// ---------------------------------------------------------------- launcher
extern "C" void kernel_launch(void* const* d_in, const int* in_sizes, int n_in,
                              void* d_out, int out_size, void* d_ws, size_t ws_size,
                              hipStream_t stream) {
    const float* x      = (const float*)d_in[0];
    const float* c      = (const float*)d_in[1];
    const float* WA_in  = (const float*)d_in[2];
    const float* bA_in  = (const float*)d_in[3];
    const float* WA_hid = (const float*)d_in[4];
    const float* bA_hid = (const float*)d_in[5];
    const float* WA_out = (const float*)d_in[6];
    const float* bA_out = (const float*)d_in[7];
    const float* WB1 = (const float*)d_in[8];
    const float* bB1 = (const float*)d_in[9];
    const float* WB2 = (const float*)d_in[10];
    const float* bB2 = (const float*)d_in[11];
    const float* WB3 = (const float*)d_in[12];
    const float* bB3 = (const float*)d_in[13];
    float* out = (float*)d_out;

    char* ws = (char*)d_ws;
    float*    y2   = (float*)(ws);                       // 128 f32      (512 B)
    float*    vA   = (float*)(ws + 512);                 // 8*128*3 f32  (12288 B)
    float*    bOut = (float*)(ws + 512 + 12288);         // 8*3 f32      (96 B)
    uint32_t* WtU  = (uint32_t*)(ws + 13056);            // 65536 dwords (256 KB), 16B aligned

    k_cnet<<<1, 128, 0, stream>>>(c, WB1, bB1, WB2, bB2, WB3, bB3, y2);
    k_vout<<<NPARTS, 384, 0, stream>>>(WA_out, bA_out, y2, vA, bOut);
    k_repack<<<256, 256, 0, stream>>>(WA_hid, WtU);
    k_main<<<(NPOINTS + MT - 1) / MT, 256, 0, stream>>>(
        x, WA_in, bA_in, bA_hid, WtU, vA, bOut, out);
}
